// MultiHeadAttention_32959579030424
// MI455X (gfx1250) — matrix operand order repaired and checked by reading, not compiled, not run on hardware
//
#include <hip/hip_runtime.h>

// ---------------------------------------------------------------------------
// MI455X (gfx1250) multi-head attention, bf16 WMMA pipeline, fp32 accumulate.
// ---------------------------------------------------------------------------

typedef __bf16 bf16;
typedef __attribute__((ext_vector_type(16))) __bf16 bf16x16;
typedef __attribute__((ext_vector_type(8)))  float  f32x8;

struct U4x2 { uint4 lo, hi; };

__device__ __forceinline__ bf16x16 load_frag(const void* p0, const void* p1) {
  U4x2 t;
  t.lo = *(const uint4*)p0;
  t.hi = *(const uint4*)p1;
  return __builtin_bit_cast(bf16x16, t);
}

#define BM 128
#define BN 128
#define BK 32
#define LDP 40   // padded LDS row stride (bf16): 80B rows, 16B-aligned chunks
#define LDF 80   // flash-attn P-tile padded stride: 160B rows

// ---------------------------------------------------------------------------
// GEMM: C[M,1024] = (A[M,1024] @ W[1024,1024]^T + bias) * outScale
// MODE 0: bf16 out, head-split [B=4, H=16, S=2048, Dh=64]     (Q, K)
// MODE 2: bf16 out, head-split transposed [B, H, Dh=64, S]     (V)
// MODE 1: fp32 out, [M, 1024]                                  (final proj)
// ---------------------------------------------------------------------------
template <int MODE>
__global__ __launch_bounds__(256)
void mha_gemm_xWT(const float* __restrict__ A, const float* __restrict__ W,
                  const float* __restrict__ bias, float* __restrict__ outF,
                  bf16* __restrict__ outH, int K, float outScale)
{
  __shared__ __align__(16) bf16 As[BM * LDP];
  __shared__ __align__(16) bf16 Bs[BN * LDP];

  const int tid  = threadIdx.x;
  const int lane = tid & 31;
  const int wave = tid >> 5;
  const int m0 = blockIdx.y * BM;
  const int n0 = blockIdx.x * BN;

  const int wm = (wave & 3) * 32;   // 4 waves along M (32 rows each)
  const int wn = (wave >> 2) * 64;  // 2 waves along N (64 cols each)

  const int r16 = lane & 15;
  const int sel = lane >> 4;

  f32x8 acc[2][4] = {};

  // staging: each thread converts 16 fp32 -> bf16 for A and for W
  const int srow = tid >> 1;            // 0..127
  const int scol = (tid & 1) * 16;      // 0 or 16

  const float* aSrc = A + (size_t)(m0 + srow) * K + scol;
  const float* wSrc = W + (size_t)(n0 + srow) * K + scol;
  bf16* aDst = As + srow * LDP + scol;
  bf16* wDst = Bs + srow * LDP + scol;

  for (int k0 = 0; k0 < K; k0 += BK) {
    __syncthreads();
    #pragma unroll
    for (int j = 0; j < 16; ++j) aDst[j] = (bf16)aSrc[j];
    #pragma unroll
    for (int j = 0; j < 16; ++j) wDst[j] = (bf16)wSrc[j];
    if (k0 + BK < K) {                      // prefetch next k-tile rows
      __builtin_prefetch(aSrc + BK, 0, 0);  // -> global_prefetch_b8
      __builtin_prefetch(wSrc + BK, 0, 0);
    }
    aSrc += BK; wSrc += BK;
    __syncthreads();

    bf16x16 af[2], bfr[4];
    #pragma unroll
    for (int i = 0; i < 2; ++i) {
      const bf16* p = As + (wm + i * 16 + r16) * LDP;
      af[i] = load_frag(p + sel * 8, p + 16 + sel * 8);     // A: lane=M
    }
    #pragma unroll
    for (int j = 0; j < 4; ++j) {
      const bf16* p = Bs + (wn + j * 16 + r16) * LDP;
      bfr[j] = load_frag(p + sel * 8, p + 16 + sel * 8);     // B: lane=N
    }
    #pragma unroll
    for (int i = 0; i < 2; ++i)
      #pragma unroll
      for (int j = 0; j < 4; ++j)
        acc[i][j] = __builtin_amdgcn_wmma_f32_16x16x32_bf16(
            false, af[i], false, bfr[j], (short)0, acc[i][j], false, false);
  }

  // epilogue: C layout -> VGPR r holds M = r + 8*sel, lane%16 = N
  #pragma unroll
  for (int i = 0; i < 2; ++i) {
    #pragma unroll
    for (int j = 0; j < 4; ++j) {
      const int n  = n0 + wn + j * 16 + r16;
      const float bv = bias[n];
      const int mbase = m0 + wm + i * 16 + 8 * sel;   // 8 consecutive rows
      if (MODE == 2) {
        // V transposed: [B,H,64,S]; 8 consecutive s -> one 16B store
        const int b = mbase >> 11, s = mbase & 2047;
        const int h = n >> 6,      d = n & 63;
        unsigned short tmp[8];
        #pragma unroll
        for (int r = 0; r < 8; ++r)
          tmp[r] = __builtin_bit_cast(unsigned short,
                     (bf16)((acc[i][j][r] + bv) * outScale));
        *(uint4*)(outH + ((size_t)(b * 16 + h) * 64 + d) * 2048 + s) =
            *(const uint4*)tmp;
      } else {
        #pragma unroll
        for (int r = 0; r < 8; ++r) {
          const int m = mbase + r;
          const float v = (acc[i][j][r] + bv) * outScale;
          if (MODE == 0) {
            const int b = m >> 11, s = m & 2047;
            const int h = n >> 6,  d = n & 63;
            outH[(((size_t)(b * 16 + h) * 2048 + s) << 6) + d] = (bf16)v;
          } else {
            outF[(size_t)m * 1024 + n] = v;
          }
        }
      }
    }
  }
}

// ---------------------------------------------------------------------------
// Flash attention: softmax(Q K^T (pre-scaled), masked) V   per (b,h).
// Grid: x = S/128 query blocks, y = B*H. 256 threads = 8 waves; each wave
// owns 16 query rows end-to-end; 64-key blocks; barrier-free (Q/K row-major
// and V pre-transposed feed WMMA fragments straight from L2; only the
// per-wave P tile round-trips LDS, ordered by in-order DS + s_wait_dscnt).
// ---------------------------------------------------------------------------
__global__ __launch_bounds__(256)
void mha_flash_attn(const bf16* __restrict__ Q,
                    const bf16* __restrict__ Kh,
                    const bf16* __restrict__ Vt,   // [B,H,64,S]
                    const int* __restrict__ mask,
                    float* __restrict__ ctx)
{
  __shared__ __align__(16) bf16 Pl[8][16 * LDF];   // per-wave P tile (16x64)

  const int tid  = threadIdx.x;
  const int lane = tid & 31;
  const int wave = tid >> 5;
  const int bh = blockIdx.y;
  const int b  = bh >> 4, h = bh & 15;
  const int q0 = blockIdx.x * 128 + wave * 16;
  const int r16 = lane & 15;
  const int sel = lane >> 4;

  const size_t headbase = (size_t)bh * 2048 * 64;   // Q/K rows
  const size_t vbase    = (size_t)bh * 64 * 2048;   // Vt rows (d-major)

  // Q A-fragments (Q was pre-scaled by 1/8 in the projection epilogue)
  bf16x16 aq[2];
  {
    const bf16* qrow = Q + headbase + (size_t)(q0 + r16) * 64;
    #pragma unroll
    for (int f = 0; f < 2; ++f)
      aq[f] = load_frag(qrow + f * 32 + sel * 8, qrow + f * 32 + 16 + sel * 8);
  }

  f32x8 o[4] = {};
  float mrow[8], lrow[8];
  #pragma unroll
  for (int r = 0; r < 8; ++r) { mrow[r] = -3.0e38f; lrow[r] = 0.0f; }

  for (int s0 = 0; s0 < 2048; s0 += 64) {
    if (s0 + 64 < 2048) {                        // prefetch next K/V blocks
      __builtin_prefetch(Kh + headbase + (size_t)(s0 + 64 + lane) * 64, 0, 0);
      __builtin_prefetch(Kh + headbase + (size_t)(s0 + 96 + lane) * 64, 0, 0);
      __builtin_prefetch(Vt + vbase + (size_t)lane        * 2048 + s0 + 64, 0, 0);
      __builtin_prefetch(Vt + vbase + (size_t)(lane + 32) * 2048 + s0 + 64, 0, 0);
    }

    // ---- scores: Q(16x64) @ K_blk(64x64)^T, four 16x16 n-tiles ------------
    f32x8 sc[4] = {};
    #pragma unroll
    for (int t = 0; t < 4; ++t) {
      const bf16* krow = Kh + headbase + (size_t)(s0 + t * 16 + r16) * 64;
      #pragma unroll
      for (int f = 0; f < 2; ++f) {
        const bf16* kp = krow + f * 32 + sel * 8;
        bf16x16 bk = load_frag(kp, kp + 16);
        sc[t] = __builtin_amdgcn_wmma_f32_16x16x32_bf16(
            false, aq[f], false, bk, (short)0, sc[t], false, false);
      }
    }

    // ---- mask + online softmax --------------------------------------------
    float p[4][8], rm[8];
    #pragma unroll
    for (int r = 0; r < 8; ++r) {
      const int qrow = q0 + r + 8 * sel;
      const int* mrowp = mask + (size_t)qrow * 2048 + s0 + r16;
      float mx = -3.0e38f;
      #pragma unroll
      for (int t = 0; t < 4; ++t) {
        float v = sc[t][r];
        if (mrowp[t * 16] == 0) v = -1e20f;
        p[t][r] = v;
        mx = fmaxf(mx, v);
      }
      rm[r] = mx;
    }
    #pragma unroll
    for (int off = 8; off >= 1; off >>= 1)
      #pragma unroll
      for (int r = 0; r < 8; ++r)
        rm[r] = fmaxf(rm[r], __shfl_xor(rm[r], off, 32));  // within 16-lane half

    float rs[8];
    #pragma unroll
    for (int r = 0; r < 8; ++r) {
      const float mn   = fmaxf(mrow[r], rm[r]);
      const float corr = __expf(mrow[r] - mn);
      mrow[r] = mn;
      bf16* pr = &Pl[wave][(r + 8 * sel) * LDF];
      float sum = 0.0f;
      #pragma unroll
      for (int t = 0; t < 4; ++t) {
        const float e = __expf(p[t][r] - mn);
        sum += e;
        pr[t * 16 + r16] = (bf16)e;              // P tile, C-layout -> LDS
      }
      rs[r] = sum;
      lrow[r] *= corr;
      #pragma unroll
      for (int dt = 0; dt < 4; ++dt) o[dt][r] *= corr;
    }
    #pragma unroll
    for (int off = 8; off >= 1; off >>= 1)
      #pragma unroll
      for (int r = 0; r < 8; ++r)
        rs[r] += __shfl_xor(rs[r], off, 32);
    #pragma unroll
    for (int r = 0; r < 8; ++r) lrow[r] += rs[r];

    // per-wave DS ops are in-order; block compiler reordering + drain DS
    asm volatile("s_wait_dscnt 0" ::: "memory");

    // ---- O += P(16x64) @ V_blk(64x64), V fragments straight from L2 -------
    #pragma unroll
    for (int kt = 0; kt < 2; ++kt) {
      bf16x16 ap;
      {
        const bf16* pr = &Pl[wave][r16 * LDF] + kt * 32;
        ap = load_frag(pr + sel * 8, pr + 16 + sel * 8);
      }
      #pragma unroll
      for (int dt = 0; dt < 4; ++dt) {
        const bf16* vp = Vt + vbase + (size_t)(dt * 16 + r16) * 2048
                         + s0 + kt * 32 + sel * 8;
        bf16x16 bv = load_frag(vp, vp + 16);
        o[dt] = __builtin_amdgcn_wmma_f32_16x16x32_bf16(
            false, ap, false, bv, (short)0, o[dt], false, false);
      }
    }
  }

  // ---- epilogue: normalize, scatter heads back to [B, S, 1024] fp32 -------
  #pragma unroll
  for (int r = 0; r < 8; ++r) {
    const float inv = 1.0f / lrow[r];
    const int srow = q0 + r + 8 * sel;
    float* dst = ctx + ((size_t)(b * 2048 + srow) * 1024) + h * 64;
    #pragma unroll
    for (int dt = 0; dt < 4; ++dt)
      dst[dt * 16 + r16] = o[dt][r] * inv;
  }
}

// ---------------------------------------------------------------------------
extern "C" void kernel_launch(void* const* d_in, const int* in_sizes, int n_in,
                              void* d_out, int out_size, void* d_ws, size_t ws_size,
                              hipStream_t stream)
{
  const float* query = (const float*)d_in[0];
  const float* key   = (const float*)d_in[1];
  const float* value = (const float*)d_in[2];
  const int*   mask  = (const int*)d_in[3];
  const float* Wq = (const float*)d_in[4];
  const float* bq = (const float*)d_in[5];
  const float* Wk = (const float*)d_in[6];
  const float* bk = (const float*)d_in[7];
  const float* Wv = (const float*)d_in[8];
  const float* bv = (const float*)d_in[9];
  const float* Wo = (const float*)d_in[10];
  const float* bo = (const float*)d_in[11];

  char* ws = (char*)d_ws;
  // bf16 Q/K row-major + V transposed in [B,H,...]: 16 MiB each (exact).
  bf16*  Qws = (bf16*)(ws);
  bf16*  Kws = (bf16*)(ws + (size_t)16 * 1024 * 1024);
  bf16*  Vws = (bf16*)(ws + (size_t)32 * 1024 * 1024);
  float* ctx = (float*)(ws + (size_t)48 * 1024 * 1024);

  dim3 blk(256);
  dim3 ggrid(1024 / BN, 8192 / BM);   // (8, 64)
  // Fold the 1/sqrt(Dh)=0.125 attention scale into the Q projection.
  mha_gemm_xWT<0><<<ggrid, blk, 0, stream>>>(query, Wq, bq, nullptr, Qws, 1024, 0.125f);
  mha_gemm_xWT<0><<<ggrid, blk, 0, stream>>>(key,   Wk, bk, nullptr, Kws, 1024, 1.0f);
  mha_gemm_xWT<2><<<ggrid, blk, 0, stream>>>(value, Wv, bv, nullptr, Vws, 1024, 1.0f);

  dim3 agrid(2048 / 128, 64);         // (16, B*H)
  mha_flash_attn<<<agrid, blk, 0, stream>>>(Qws, Kws, Vws, mask, ctx);

  mha_gemm_xWT<1><<<ggrid, blk, 0, stream>>>(ctx, Wo, bo, (float*)d_out, nullptr, 1024, 1.0f);
}
